// GAPTGN_18339510354211
// MI455X (gfx1250) — matrix operand order repaired and checked by reading, not compiled, not run on hardware
//
#include <hip/hip_runtime.h>
#include <hip/hip_bf16.h>

// ---------------------------------------------------------------------------
// Problem constants (match reference)
// ---------------------------------------------------------------------------
#define TGN_H    128     // memory / hidden width
#define TGN_EF   64      // raw message width
#define TGN_DMSG 448     // 128 + 128 + 64 + 128
#define TGN_SL   50      // price sequence length

typedef __attribute__((ext_vector_type(16))) _Float16 v16h;
typedef __attribute__((ext_vector_type(8)))  float    v8f;

// ---------------------------------------------------------------------------
// WMMA fragment helpers (contiguous k order shared by A and B, 16-bit, wave32)
//
// A (16x32, MxK), row-major f16 tile in LDS with leading dim ldk:
//   lanes 0-15 : row M=lane,    elems 0..15 -> K=k0..k0+15
//   lanes 16-31: row M=lane-16, elems 0..15 -> K=k0+16..k0+31
// ---------------------------------------------------------------------------
__device__ __forceinline__ v16h ldsA_frag(const _Float16* tile, int ldk,
                                          int k0, int lane) {
  const int row = lane & 15;
  const int kb  = k0 + ((lane & 16) ? 16 : 0);
  const _Float16* p = tile + row * ldk + kb;
  v16h a;
#pragma unroll
  for (int i = 0; i < 8; ++i) a[i] = p[i];
#pragma unroll
  for (int i = 0; i < 8; ++i) a[i + 8] = p[i + 8];
  return a;
}

// B (32x16, KxN) fragments pre-packed in global memory: for fragment f,
// lane l holds 16 contiguous f16 (32B load) with
//   n = (l&15) + 16*ntile,  k_local = ((l&16)?16:0) + elem
__device__ __forceinline__ v16h ldg_B_frag(const _Float16* packed, int frag,
                                           int lane) {
  const _Float16* p = packed + (((size_t)frag << 5) + lane) * 16;
  v16h b;
#pragma unroll
  for (int i = 0; i < 16; ++i) b[i] = p[i];
  return b;
}

__device__ __forceinline__ v8f wmma_f16(v16h a, v16h b, v8f c) {
  return __builtin_amdgcn_wmma_f32_16x16x32_f16(false, a, false, b,
                                                (short)0, c, false, false);
}

__device__ __forceinline__ float sigmoidf_(float x) {
  return 1.0f / (1.0f + __expf(-x));
}

// ---------------------------------------------------------------------------
// Weight packer: W[Nout,K] row-major f32 -> f16 WMMA-B fragments.
// fragment index f = ntile*(K/32) + ktile ; element order matches ldg_B_frag.
// ---------------------------------------------------------------------------
__global__ void k_pack_b(const float* __restrict__ W, _Float16* __restrict__ out,
                         int Nout, int K) {
  const int nK = K >> 5;
  const size_t total = (size_t)Nout * (size_t)K;
  size_t idx = (size_t)blockIdx.x * blockDim.x + threadIdx.x;
  if (idx >= total) return;
  const int e = (int)(idx & 15);
  const int l = (int)((idx >> 4) & 31);
  const int f = (int)(idx >> 9);
  const int kt = f % nK;
  const int nt = f / nK;
  const int n = nt * 16 + (l & 15);
  const int k = kt * 32 + ((l & 16) ? 16 : 0) + e;
  out[idx] = (_Float16)W[(size_t)n * K + k];
}

// ---------------------------------------------------------------------------
// LastAggregator winner selection
// ---------------------------------------------------------------------------
__global__ void k_init_winner(int* __restrict__ maxt, int* __restrict__ win, int N) {
  int i = blockIdx.x * blockDim.x + threadIdx.x;
  if (i < N) { maxt[i] = -1; win[i] = -1; }
}

__global__ void k_scatter_maxt(const int* __restrict__ src, const int* __restrict__ dst,
                               const int* __restrict__ t, int* __restrict__ maxt, int E) {
  int i = blockIdx.x * blockDim.x + threadIdx.x;
  if (i >= 2 * E) return;
  const int er = (i < E) ? i : (i - E);
  const int node = (i < E) ? src[er] : dst[er];
  atomicMax(&maxt[node], t[er]);
}

__global__ void k_scatter_win(const int* __restrict__ src, const int* __restrict__ dst,
                              const int* __restrict__ t, const int* __restrict__ maxt,
                              int* __restrict__ win, int E) {
  int i = blockIdx.x * blockDim.x + threadIdx.x;
  if (i >= 2 * E) return;
  const int er = (i < E) ? i : (i - E);
  const int node = (i < E) ? src[er] : dst[er];
  if (t[er] == maxt[node]) atomicMax(&win[node], i);
}

// ---------------------------------------------------------------------------
// GRU message+update kernel, v2: 32-row tile, 8 waves.
// Wave w owns the ntile triple (w, w+8, w+16) => columns (j, 128+j, 256+j)
// share one lane: r/z/n gate math is fully register-resident (no G in LDS).
// G1 = m @ Wih^T (K=448), G2 = h @ Whh^T (K=128, A = cols 0..127 of m tile).
// ---------------------------------------------------------------------------
__global__ __launch_bounds__(256) void k_gru(
    const int* __restrict__ src, const int* __restrict__ dst,
    const int* __restrict__ t, const float* __restrict__ msg,
    const float* __restrict__ memory, const int* __restrict__ last_update,
    const float* __restrict__ time_w, const float* __restrict__ time_b,
    const _Float16* __restrict__ wih_p, const _Float16* __restrict__ whh_p,
    const float* __restrict__ bih, const float* __restrict__ bhh,
    float* __restrict__ hnew, int E) {
  __shared__ _Float16 sM[32 * 456];      // message tile (ldk=456, padded)
  __shared__ float    sHf[32 * 128];     // z_src f32 (final blend)

  const int tid  = threadIdx.x;
  const int lane = tid & 31;
  const int wv   = tid >> 5;             // 0..7
  const int row0 = blockIdx.x * 32;

  // ---- build the message tile: m = [z_src | z_dst | raw | t_enc] ----
  {
    const int r   = tid >> 3;            // 0..31
    const int sub = tid & 7;             // 0..7
    const int gr  = row0 + r;
    const bool fs = gr < E;
    const int er    = fs ? gr : (gr - E);
    const int node  = fs ? src[er] : dst[er];
    const int other = fs ? dst[er] : src[er];
    const float relt = (float)(t[er] - last_update[node]);
    const float* zs = memory + (size_t)node  * TGN_H;
    const float* zd = memory + (size_t)other * TGN_H;
    const float* mr = msg + (size_t)er * TGN_EF;
    for (int j = sub; j < TGN_H; j += 8) {
      const float v = zs[j];
      sM [r * 456 + j]       = (_Float16)v;
      sHf[r * 128 + j]       = v;
      sM [r * 456 + 128 + j] = (_Float16)zd[j];
      sM [r * 456 + 320 + j] = (_Float16)cosf(relt * time_w[j] + time_b[j]);
    }
    for (int j = sub; j < TGN_EF; j += 8)
      sM[r * 456 + 256 + j] = (_Float16)mr[j];
  }
  __syncthreads();

  // ---- G1 = m @ Wih^T : 3 ntiles x 2 mtiles of register accumulators ----
  v8f g1[6] = {};
  for (int kt = 0; kt < 14; ++kt) {
    const v16h a0 = ldsA_frag(sM,            456, kt * 32, lane);
    const v16h a1 = ldsA_frag(sM + 16 * 456, 456, kt * 32, lane);
    if (kt + 1 < 14)  // stream-ahead hint for next k-chunk of B (this triple)
      __builtin_prefetch(wih_p + (((size_t)(wv * 14 + kt + 1)) << 9), 0, 1);
#pragma unroll
    for (int c = 0; c < 3; ++c) {
      const v16h b = ldg_B_frag(wih_p, (wv + 8 * c) * 14 + kt, lane);
      g1[c * 2 + 0] = wmma_f16(a0, b, g1[c * 2 + 0]);
      g1[c * 2 + 1] = wmma_f16(a1, b, g1[c * 2 + 1]);
    }
  }

  // ---- G2 = h @ Whh^T : A is cols 0..127 of the message tile ----
  v8f g2[6] = {};
#pragma unroll
  for (int kt = 0; kt < 4; ++kt) {
    const v16h a0 = ldsA_frag(sM,            456, kt * 32, lane);
    const v16h a1 = ldsA_frag(sM + 16 * 456, 456, kt * 32, lane);
#pragma unroll
    for (int c = 0; c < 3; ++c) {
      const v16h b = ldg_B_frag(whh_p, (wv + 8 * c) * 4 + kt, lane);
      g2[c * 2 + 0] = wmma_f16(a0, b, g2[c * 2 + 0]);
      g2[c * 2 + 1] = wmma_f16(a1, b, g2[c * 2 + 1]);
    }
  }

  // ---- fused GRU gates, all in registers ----
  const int j  = wv * 16 + (lane & 15);   // hidden index 0..127
  const int mb = (lane & 16) ? 8 : 0;
  const float br_i = bih[j],       br_h = bhh[j];
  const float bz_i = bih[128 + j], bz_h = bhh[128 + j];
  const float bn_i = bih[256 + j], bn_h = bhh[256 + j];
#pragma unroll
  for (int mt = 0; mt < 2; ++mt) {
#pragma unroll
    for (int i = 0; i < 8; ++i) {
      const int m = mt * 16 + mb + i;
      const float r  = sigmoidf_(g1[0 + mt][i] + g2[0 + mt][i] + br_i + br_h);
      const float z  = sigmoidf_(g1[2 + mt][i] + g2[2 + mt][i] + bz_i + bz_h);
      const float nn = tanhf(g1[4 + mt][i] + bn_i + r * (g2[4 + mt][i] + bn_h));
      const float hp = sHf[m * 128 + j];
      hnew[(size_t)(row0 + m) * TGN_H + j] = (1.0f - z) * nn + z * hp;
    }
  }
}

// ---------------------------------------------------------------------------
// Price LSTM (hidden 32, 50 steps) + projection -> price_emb[E,128]
// Recurrent B fragments are register-resident across the whole time scan.
// ---------------------------------------------------------------------------
__global__ __launch_bounds__(128) void k_lstm(
    const float* __restrict__ price_seq, const float* __restrict__ lstm_Wih,
    const float* __restrict__ lstm_bih, const float* __restrict__ lstm_bhh,
    const _Float16* __restrict__ lwhh_p, const _Float16* __restrict__ proj_p,
    const float* __restrict__ proj_b, float* __restrict__ pemb, int E) {
  __shared__ _Float16 sH[16 * 32];
  __shared__ float    sHc[16 * 32];   // cell state
  __shared__ float    sG[16 * 128];
  __shared__ float    sX[16 * TGN_SL];
  __shared__ float    sWih[128];
  __shared__ float    sB[128];

  const int tid  = threadIdx.x;
  const int lane = tid & 31;
  const int wv   = tid >> 5;
  const int row0 = blockIdx.x * 16;

  sWih[tid] = lstm_Wih[tid];
  sB[tid]   = lstm_bih[tid] + lstm_bhh[tid];
  for (int q = tid; q < 16 * TGN_SL; q += 128) {
    const int r = q / TGN_SL, s = q % TGN_SL;
    sX[q] = price_seq[(size_t)(row0 + r) * TGN_SL + s];
  }
  for (int q = tid; q < 16 * 32; q += 128) { sH[q] = (_Float16)0.0f; sHc[q] = 0.0f; }

  // recurrent weights: wave wv owns ntiles 2wv, 2wv+1; load B frags ONCE
  v16h bw[2];
#pragma unroll
  for (int c = 0; c < 2; ++c) bw[c] = ldg_B_frag(lwhh_p, wv * 2 + c, lane);
  __syncthreads();

  for (int s = 0; s < TGN_SL; ++s) {
    // g_rec = h @ Whh^T  (K=32: one WMMA chunk per ntile)
#pragma unroll
    for (int c = 0; c < 2; ++c) {
      const v16h a = ldsA_frag(sH, 32, 0, lane);
      v8f acc = {};
      acc = wmma_f16(a, bw[c], acc);
      const int n  = (wv * 2 + c) * 16 + (lane & 15);
      const int mb = (lane & 16) ? 8 : 0;
#pragma unroll
      for (int i = 0; i < 8; ++i) sG[(mb + i) * 128 + n] = acc[i];
    }
    __syncthreads();
    // fused gates
    for (int q = tid; q < 16 * 32; q += 128) {
      const int m = q >> 5, jj = q & 31;
      const float x  = sX[m * TGN_SL + s];
      const float gi = sG[m * 128 + jj]      + x * sWih[jj]      + sB[jj];
      const float gf = sG[m * 128 + 32 + jj] + x * sWih[32 + jj] + sB[32 + jj];
      const float gg = sG[m * 128 + 64 + jj] + x * sWih[64 + jj] + sB[64 + jj];
      const float go = sG[m * 128 + 96 + jj] + x * sWih[96 + jj] + sB[96 + jj];
      const float c  = sigmoidf_(gf) * sHc[q] + sigmoidf_(gi) * tanhf(gg);
      const float h  = sigmoidf_(go) * tanhf(c);
      sHc[q] = c;
      sH[q]  = (_Float16)h;
    }
    __syncthreads();
  }

  // price_emb = h_n @ proj^T + proj_b
  for (int nt = wv; nt < 8; nt += 4) {
    const v16h a = ldsA_frag(sH, 32, 0, lane);
    const v16h b = ldg_B_frag(proj_p, nt, lane);
    v8f acc = {};
    acc = wmma_f16(a, b, acc);
    const int n  = nt * 16 + (lane & 15);
    const int mb = (lane & 16) ? 8 : 0;
    const float pb = proj_b[n];
#pragma unroll
    for (int i = 0; i < 8; ++i)
      pemb[(size_t)(row0 + mb + i) * TGN_H + n] = acc[i] + pb;
  }
}

// ---------------------------------------------------------------------------
// Head: encoders + combine + predictor MLP -> out[E]
// ---------------------------------------------------------------------------
__global__ __launch_bounds__(128) void k_head(
    const int* __restrict__ src, const int* __restrict__ dst,
    const int* __restrict__ win, const float* __restrict__ hnew,
    const float* __restrict__ x_pol, const float* __restrict__ x_comp,
    const _Float16* __restrict__ pol_p, const float* __restrict__ pol_b,
    const _Float16* __restrict__ comp_p, const float* __restrict__ comp_b,
    const float* __restrict__ pemb, const _Float16* __restrict__ p1_p,
    const float* __restrict__ p1_b, const float* __restrict__ p2_W,
    const float* __restrict__ p2_b, float* __restrict__ out, int E) {
  __shared__ _Float16 sA[16 * 64];       // x_pol then x_comp tile
  __shared__ float    sMs[16 * 128];     // mem_src
  __shared__ float    sMd[16 * 128];     // mem_dst
  __shared__ float    sPe[16 * 128];     // price_emb
  __shared__ _Float16 sComb[16 * 392];   // combined [16 x 384], ldk=392
  __shared__ float    sHid[16 * 64];

  const int tid  = threadIdx.x;
  const int lane = tid & 31;
  const int wv   = tid >> 5;
  const int row0 = blockIdx.x * 16;

  {
    const int r = tid >> 3, sub = tid & 7;
    const int gr = row0 + r;
    const int wr = win[src[gr]];   // every touched node has exactly one winner
    const int wd = win[dst[gr]];
    const float* ps = hnew + (size_t)wr * TGN_H;
    const float* pd = hnew + (size_t)wd * TGN_H;
    const float* pp = pemb + (size_t)gr * TGN_H;
    for (int j = sub; j < TGN_H; j += 8) {
      sMs[r * 128 + j] = ps[j];
      sMd[r * 128 + j] = pd[j];
      const float pe = pp[j];
      sPe[r * 128 + j] = pe;
      sComb[r * 392 + 256 + j] = (_Float16)pe;
    }
    for (int j = sub; j < 64; j += 8)
      sA[r * 64 + j] = (_Float16)x_pol[(size_t)gr * 64 + j];
  }
  __syncthreads();

  // pol_ctx = mem_src + x_pol @ pol_W^T + pol_b  -> comb[:,0:128]
  for (int nt = wv; nt < 8; nt += 4) {
    v8f acc = {};
#pragma unroll
    for (int kt = 0; kt < 2; ++kt) {
      const v16h a = ldsA_frag(sA, 64, kt * 32, lane);
      const v16h b = ldg_B_frag(pol_p, nt * 2 + kt, lane);
      acc = wmma_f16(a, b, acc);
    }
    const int n  = nt * 16 + (lane & 15);
    const int mb = (lane & 16) ? 8 : 0;
    const float pb = pol_b[n];
#pragma unroll
    for (int i = 0; i < 8; ++i) {
      const int m = mb + i;
      sComb[m * 392 + n] = (_Float16)(sMs[m * 128 + n] + acc[i] + pb);
    }
  }
  __syncthreads();

  {  // swap in x_comp
    const int r = tid >> 3, sub = tid & 7;
    const int gr = row0 + r;
    for (int j = sub; j < 64; j += 8)
      sA[r * 64 + j] = (_Float16)x_comp[(size_t)gr * 64 + j];
  }
  __syncthreads();

  // comp_ctx = mem_dst + comp_emb + price_emb -> comb[:,128:256]
  for (int nt = wv; nt < 8; nt += 4) {
    v8f acc = {};
#pragma unroll
    for (int kt = 0; kt < 2; ++kt) {
      const v16h a = ldsA_frag(sA, 64, kt * 32, lane);
      const v16h b = ldg_B_frag(comp_p, nt * 2 + kt, lane);
      acc = wmma_f16(a, b, acc);
    }
    const int n  = nt * 16 + (lane & 15);
    const int mb = (lane & 16) ? 8 : 0;
    const float cb = comp_b[n];
#pragma unroll
    for (int i = 0; i < 8; ++i) {
      const int m = mb + i;
      sComb[m * 392 + 128 + n] =
          (_Float16)(sMd[m * 128 + n] + acc[i] + cb + sPe[m * 128 + n]);
    }
  }
  __syncthreads();

  // hid = relu(combined @ p1^T + b)  (N=64: 1 ntile/wave, K=384: 12 chunks)
  {
    const int nt = wv;
    v8f acc = {};
    for (int kt = 0; kt < 12; ++kt) {
      const v16h a = ldsA_frag(sComb, 392, kt * 32, lane);
      const v16h b = ldg_B_frag(p1_p, nt * 12 + kt, lane);
      acc = wmma_f16(a, b, acc);
    }
    const int n  = nt * 16 + (lane & 15);
    const int mb = (lane & 16) ? 8 : 0;
    const float pb = p1_b[n];
#pragma unroll
    for (int i = 0; i < 8; ++i)
      sHid[(mb + i) * 64 + n] = fmaxf(acc[i] + pb, 0.0f);
  }
  __syncthreads();

  // out = hid @ p2^T + b  (64-dot per row)
  if (tid < 16) {
    float s = p2_b[0];
#pragma unroll 8
    for (int j = 0; j < 64; ++j) s += sHid[tid * 64 + j] * p2_W[j];
    out[row0 + tid] = s;
  }
}

// ---------------------------------------------------------------------------
// Launch
// ---------------------------------------------------------------------------
extern "C" void kernel_launch(void* const* d_in, const int* in_sizes, int n_in,
                              void* d_out, int out_size, void* d_ws, size_t ws_size,
                              hipStream_t stream) {
  const int E = in_sizes[0];       // 16384
  const int N = in_sizes[9];       // 200000 (last_update)

  const int*   src   = (const int*)d_in[0];
  const int*   dst   = (const int*)d_in[1];
  const int*   t     = (const int*)d_in[2];
  const float* msg   = (const float*)d_in[3];
  const float* pseq  = (const float*)d_in[4];
  const float* x_pol = (const float*)d_in[6];
  const float* x_cmp = (const float*)d_in[7];
  const float* mem   = (const float*)d_in[8];
  const int*   lupd  = (const int*)d_in[9];
  const float* tw    = (const float*)d_in[10];
  const float* tb    = (const float*)d_in[11];
  const float* gWih  = (const float*)d_in[12];
  const float* gWhh  = (const float*)d_in[13];
  const float* gbih  = (const float*)d_in[14];
  const float* gbhh  = (const float*)d_in[15];
  const float* polW  = (const float*)d_in[16];
  const float* polB  = (const float*)d_in[17];
  const float* cmpW  = (const float*)d_in[18];
  const float* cmpB  = (const float*)d_in[19];
  const float* lWih  = (const float*)d_in[20];
  const float* lWhh  = (const float*)d_in[21];
  const float* lbih  = (const float*)d_in[22];
  const float* lbhh  = (const float*)d_in[23];
  const float* prW   = (const float*)d_in[24];
  const float* prB   = (const float*)d_in[25];
  const float* p1W   = (const float*)d_in[26];
  const float* p1B   = (const float*)d_in[27];
  const float* p2W   = (const float*)d_in[28];
  const float* p2B   = (const float*)d_in[29];
  float* out = (float*)d_out;

  // workspace carve-up (256B aligned)
  char* ws = (char*)d_ws;
  size_t off = 0;
  auto carve = [&](size_t bytes) {
    void* p = ws + off;
    off = (off + bytes + 255) & ~(size_t)255;
    return p;
  };
  _Float16* wih_p  = (_Float16*)carve((size_t)384 * 448 * 2);
  _Float16* whh_p  = (_Float16*)carve((size_t)384 * 128 * 2);
  _Float16* pol_p  = (_Float16*)carve((size_t)128 * 64 * 2);
  _Float16* cmp_p  = (_Float16*)carve((size_t)128 * 64 * 2);
  _Float16* lwhh_p = (_Float16*)carve((size_t)128 * 32 * 2);
  _Float16* proj_p = (_Float16*)carve((size_t)128 * 32 * 2);
  _Float16* p1_p   = (_Float16*)carve((size_t)64 * 384 * 2);
  int*   maxt = (int*)carve((size_t)N * 4);
  int*   win  = (int*)carve((size_t)N * 4);
  float* hnew = (float*)carve((size_t)2 * E * TGN_H * 4);
  float* pemb = (float*)carve((size_t)E * TGN_H * 4);
  (void)ws_size; (void)n_in; (void)out_size;

  // 1) pack weights into WMMA-B fragment layouts
  auto pack = [&](const float* W, _Float16* dstp, int Nout, int K) {
    const int total = Nout * K;
    k_pack_b<<<(total + 255) / 256, 256, 0, stream>>>(W, dstp, Nout, K);
  };
  pack(gWih, wih_p, 384, 448);
  pack(gWhh, whh_p, 384, 128);
  pack(polW, pol_p, 128, 64);
  pack(cmpW, cmp_p, 128, 64);
  pack(lWhh, lwhh_p, 128, 32);
  pack(prW,  proj_p, 128, 32);
  pack(p1W,  p1_p,  64, 384);

  // 2) LastAggregator winners
  k_init_winner<<<(N + 255) / 256, 256, 0, stream>>>(maxt, win, N);
  k_scatter_maxt<<<(2 * E + 255) / 256, 256, 0, stream>>>(src, dst, t, maxt, E);
  k_scatter_win<<<(2 * E + 255) / 256, 256, 0, stream>>>(src, dst, t, maxt, win, E);

  // 3) GRU memory update (h_new for all 2E message rows), 32 rows/block
  k_gru<<<(2 * E) / 32, 256, 0, stream>>>(src, dst, t, msg, mem, lupd, tw, tb,
                                          wih_p, whh_p, gbih, gbhh, hnew, E);

  // 4) price LSTM + projection
  k_lstm<<<E / 16, 128, 0, stream>>>(pseq, lWih, lbih, lbhh, lwhh_p, proj_p,
                                     prB, pemb, E);

  // 5) encoders + combine + predictor MLP
  k_head<<<E / 16, 128, 0, stream>>>(src, dst, win, hnew, x_pol, x_cmp,
                                     pol_p, polB, cmp_p, cmpB, pemb,
                                     p1_p, p1B, p2W, p2B, out, E);
}
